// GRUSequence_4088808866387
// MI455X (gfx1250) — matrix operand order repaired and checked by reading, not compiled, not run on hardware
//
#include <hip/hip_runtime.h>
#include <hip/hip_bf16.h>

// ---------------------------------------------------------------------------
// GRU sequence (2 layers + RMSNorm) for MI455X / gfx1250.
//   * bf16 WMMA (V_WMMA_F32_16X16X32_BF16) for all GEMMs, f32 accumulate.
//   * Input gates precomputed as two large batched GEMMs (43% of FLOPs off
//     the serial path).
//   * Recurrent gh GEMM per step: M=64, weights L2-resident in bf16; shared
//     B tile staged through LDS with gfx1250 async-to-LDS ops (ASYNCcnt).
//   * Big GEMM: register ping-pong double buffering so WMMA overlaps loads.
// ---------------------------------------------------------------------------

typedef __attribute__((ext_vector_type(16))) __bf16 v16bf;
typedef __attribute__((ext_vector_type(8)))  float  v8f;
typedef int v4i __attribute__((vector_size(16)));
typedef unsigned short u16;
typedef unsigned int   u32;
typedef unsigned long long u64b;

#define AS1 __attribute__((address_space(1)))
#define AS3 __attribute__((address_space(3)))

#if defined(__has_builtin)
#if __has_builtin(__builtin_amdgcn_global_load_async_to_lds_b128) && \
    __has_builtin(__builtin_amdgcn_s_wait_asynccnt)
#define GRU_HAVE_ASYNC_LDS 1
#endif
#endif
#ifndef GRU_HAVE_ASYNC_LDS
#define GRU_HAVE_ASYNC_LDS 0
#endif

#if GRU_HAVE_ASYNC_LDS
#define ASYNC_WAIT(n) __builtin_amdgcn_s_wait_asynccnt(n)
#else
#define ASYNC_WAIT(n)
#endif

union Frag {
    v16bf v;
    uint4 q[2];
};

__device__ __forceinline__ u16 f2bf(float f) {
    union { float f; u32 u; } c;
    c.f = f;
    u32 r = c.u + 0x7FFFu + ((c.u >> 16) & 1u);   // round-to-nearest-even
    return (u16)(r >> 16);
}

// ---------------------------------------------------------------------------
// Big GEMM: C[M,N] = A[M,K](bf16) @ W[N,K]^T(bf16) + bias[N]  (f32 out)
// Block: 128 thr = 4 waves; block tile 64(M) x 64(N); wave w owns rows
// [64*bx + 16w, +16) and ALL 64 block columns -> the 4 waves issue identical
// B loads (merged in WGP$). Register ping-pong over 32-wide K chunks so the
// load clause for chunk k+1 is in flight while WMMAs consume chunk k.
// Grid: (M/64, N/64). K multiple of 64. EXEC all-ones everywhere.
// ---------------------------------------------------------------------------
__global__ void gemm_bf16_wmma(const u16* __restrict__ A,
                               const u16* __restrict__ W,
                               const float* __restrict__ bias,
                               float* __restrict__ C,
                               int M, int N, int K) {
    const int lane   = threadIdx.x & 31;
    const int wave   = threadIdx.x >> 5;
    const int l15    = lane & 15;
    const int half   = lane >> 4;
    const int tile_m = (blockIdx.x << 6) + (wave << 4);
    const int tile_n = blockIdx.y << 6;

    // A fragment: lane holds row (tile_m + l15), 16 contiguous K at half*16.
    const u16* Aptr = A + (size_t)(tile_m + l15) * K + (half << 4);
    // B fragment: lane holds column n = tile_n + s*16 + l15, 16 contiguous K
    // starting at half*16 (gh = h @ W^T: B column n is W row n -> no transpose).
    const int b_kb = half << 4;
    const u16* Wp0 = W + (size_t)(tile_n + l15) * K + b_kb;
    const size_t wstride = (size_t)16 * K;       // 16 rows of W per subtile

    v8f acc[4] = {v8f{}, v8f{}, v8f{}, v8f{}};

    auto loadfrag = [&](Frag& a, Frag (&b)[4], int k) {
        a.q[0] = *(const uint4*)(Aptr + k);
        a.q[1] = *(const uint4*)(Aptr + k + 8);
#pragma unroll
        for (int s = 0; s < 4; ++s) {
            const u16* p = Wp0 + (size_t)s * wstride + k;
            b[s].q[0] = *(const uint4*)(p);
            b[s].q[1] = *(const uint4*)(p + 8);
        }
    };
    auto domma = [&](Frag& a, Frag (&b)[4]) {
#pragma unroll
        for (int s = 0; s < 4; ++s)
            acc[s] = __builtin_amdgcn_wmma_f32_16x16x32_bf16(
                false, a.v, false, b[s].v, (short)0, acc[s], false, false);
    };

    Frag a0, a1, b0[4], b1[4];
    loadfrag(a0, b0, 0);
    int k0 = 0;
    for (; k0 < K - 64; k0 += 64) {
        loadfrag(a1, b1, k0 + 32);   // in flight while chunk k0 computes
        domma(a0, b0);
        loadfrag(a0, b0, k0 + 64);
        domma(a1, b1);
    }
    loadfrag(a1, b1, K - 32);
    domma(a0, b0);
    domma(a1, b1);

    // C/D layout: VGPR i -> M = i + 8*half, N = l15 (ISA 7.12.2)
    const int m_half = half << 3;
#pragma unroll
    for (int s = 0; s < 4; ++s) {
        const int n   = tile_n + (s << 4) + l15;
        const float bn = bias[n];
#pragma unroll
        for (int i = 0; i < 8; ++i) {
            const int m = tile_m + m_half + i;
            C[(size_t)m * N + n] = acc[s][i] + bn;
        }
    }
}

// ---------------------------------------------------------------------------
// Recurrent gh GEMM: M = 64 fixed. C[64,N] = A[64,K] @ W[N,K]^T + bias.
// One block covers all 64 rows (wave w -> rows 16w..) and 64 columns; the B
// tile (64 cols x 32 K, 4 KB) is shared by all 4 waves, staged through LDS
// with async-to-LDS copies (double-buffered, ASYNCcnt-tracked). Per step this
// reads each weight exactly once (~6 MB bf16, L2-resident).
// Grid: (N/64). K multiple of 32.
// ---------------------------------------------------------------------------
__global__ void gru_hh_gemm(const u16* __restrict__ A,
                            const u16* __restrict__ W,
                            const float* __restrict__ bias,
                            float* __restrict__ C,
                            int N, int K) {
    constexpr int STRIDE = 40;                    // u16; 80 B rows: 16B-aligned,
    __shared__ u16 btile[2][64 * STRIDE];         // conflict-free ds_load_b128
    const int lane   = threadIdx.x & 31;
    const int wave   = threadIdx.x >> 5;
    const int l15    = lane & 15;
    const int half   = lane >> 4;
    const int tile_n = blockIdx.x << 6;
    const int tile_m = wave << 4;

    const u16* Aptr = A + (size_t)(tile_m + l15) * K + (half << 4);
    const int b_kb = half << 4;

    // Stage one 64x32 B tile into LDS: 256 x 16B transfers, 2 per thread.
    auto stage = [&](int buf, int k) {
#pragma unroll
        for (int c = threadIdx.x; c < 256; c += 128) {
            const int r = c >> 2, seg = c & 3;
            const u16* src = W + (size_t)(tile_n + r) * K + k + (seg << 3);
            u16* dst = &btile[buf][r * STRIDE + (seg << 3)];
#if GRU_HAVE_ASYNC_LDS
            __builtin_amdgcn_global_load_async_to_lds_b128(
                (AS1 v4i*)(u64b)src, (AS3 v4i*)(u32)(u64b)dst, 0, 0);
#else
            *(uint4*)dst = *(const uint4*)src;
#endif
        }
    };

    v8f acc[4] = {v8f{}, v8f{}, v8f{}, v8f{}};

    auto compute = [&](int buf, int k) {
        Frag a;
        a.q[0] = *(const uint4*)(Aptr + k);
        a.q[1] = *(const uint4*)(Aptr + k + 8);
#pragma unroll
        for (int s = 0; s < 4; ++s) {
            Frag b;
            const u16* p = &btile[buf][((s << 4) + l15) * STRIDE + b_kb];
            b.q[0] = *(const uint4*)(p);
            b.q[1] = *(const uint4*)(p + 8);
            acc[s] = __builtin_amdgcn_wmma_f32_16x16x32_bf16(
                false, a.v, false, b.v, (short)0, acc[s], false, false);
        }
    };

    const int nchunk = K >> 5;
    stage(0, 0);
    for (int i = 0; i < nchunk - 1; ++i) {
        const int buf = i & 1;
        __syncthreads();                 // prior readers done before restaging
        stage(buf ^ 1, (i + 1) << 5);    // prefetch next chunk (2 async ops)
        ASYNC_WAIT(2);                   // my chunk-i copies have landed
        __syncthreads();                 // everyone's chunk-i copies landed
        compute(buf, i << 5);
    }
    ASYNC_WAIT(0);
    __syncthreads();
    compute((nchunk - 1) & 1, (nchunk - 1) << 5);

    const int m_half = half << 3;
#pragma unroll
    for (int s = 0; s < 4; ++s) {
        const int n   = tile_n + (s << 4) + l15;
        const float bn = bias[n];
#pragma unroll
        for (int i = 0; i < 8; ++i) {
            const int m = tile_m + m_half + i;
            C[(size_t)m * N + n] = acc[s][i] + bn;
        }
    }
}

// ---------------------------------------------------------------------------
// Fused GRU gate + RMSNorm for one timestep. One block (256 thr) per batch row.
// ---------------------------------------------------------------------------
__global__ void gru_gate_rmsnorm(const float* __restrict__ gi, size_t gi_bstride,
                                 const float* __restrict__ gh,
                                 const float* __restrict__ hprev,
                                 const float* __restrict__ g,
                                 float* __restrict__ hf_out,
                                 u16*   __restrict__ hbf_out,
                                 float* __restrict__ seq_f32,   // nullable
                                 u16*   __restrict__ seq_bf,    // nullable
                                 size_t seq_bstride, int H) {
    const int b = blockIdx.x;
    const float* gib = gi + (size_t)b * gi_bstride;
    const float* ghb = gh + (size_t)b * 3 * H;
    const float* hb  = hprev + (size_t)b * H;

    float vals[4];
    float ss = 0.0f;
#pragma unroll
    for (int u = 0; u < 4; ++u) {
        const int j = threadIdx.x + (u << 8);
        const float ir = gib[j], iz = gib[H + j], inn = gib[2 * H + j];
        const float hr = ghb[j], hz = ghb[H + j], hnn = ghb[2 * H + j];
        const float r  = 1.0f / (1.0f + __expf(-(ir + hr)));
        const float z  = 1.0f / (1.0f + __expf(-(iz + hz)));
        const float n  = tanhf(inn + r * hnn);
        const float hv = (1.0f - z) * n + z * hb[j];
        vals[u] = hv;
        ss += hv * hv;
    }
#pragma unroll
    for (int off = 16; off > 0; off >>= 1) ss += __shfl_xor(ss, off, 32);
    __shared__ float red[8];
    if ((threadIdx.x & 31) == 0) red[threadIdx.x >> 5] = ss;
    __syncthreads();
    float tot = 0.0f;
#pragma unroll
    for (int w = 0; w < 8; ++w) tot += red[w];
    const float scale = rsqrtf(tot / (float)H + 1e-6f);

#pragma unroll
    for (int u = 0; u < 4; ++u) {
        const int j = threadIdx.x + (u << 8);
        const float o = vals[u] * scale * g[j];
        hf_out [(size_t)b * H + j] = o;
        hbf_out[(size_t)b * H + j] = f2bf(o);
        if (seq_f32) seq_f32[(size_t)b * seq_bstride + j] = o;
        if (seq_bf)  seq_bf [(size_t)b * seq_bstride + j] = f2bf(o);
    }
}

__global__ void convert_bf16(const float* __restrict__ src, u16* __restrict__ dst,
                             size_t n) {
    size_t i = (size_t)blockIdx.x * blockDim.x + threadIdx.x;
    if (i < n) dst[i] = f2bf(src[i]);
}

__global__ void init_state(const float* __restrict__ src, float* __restrict__ hf,
                           u16* __restrict__ hbf, int n) {
    int i = blockIdx.x * blockDim.x + threadIdx.x;
    if (i < n) {
        const float v = src[i];
        hf[i]  = v;
        hbf[i] = f2bf(v);
    }
}

// ---------------------------------------------------------------------------
extern "C" void kernel_launch(void* const* d_in, const int* in_sizes, int n_in,
                              void* d_out, int out_size, void* d_ws, size_t ws_size,
                              hipStream_t stream) {
    (void)in_sizes; (void)n_in; (void)out_size; (void)ws_size;

    constexpr int B = 64, T = 512, DIN = 512, H = 1024;
    constexpr int H3 = 3 * H;

    const float* x     = (const float*)d_in[0];
    const float* h0    = (const float*)d_in[1];
    const float* w_ih0 = (const float*)d_in[2];
    const float* w_hh0 = (const float*)d_in[3];
    const float* b_ih0 = (const float*)d_in[4];
    const float* b_hh0 = (const float*)d_in[5];
    const float* g0    = (const float*)d_in[6];
    const float* w_ih1 = (const float*)d_in[7];
    const float* w_hh1 = (const float*)d_in[8];
    const float* b_ih1 = (const float*)d_in[9];
    const float* b_hh1 = (const float*)d_in[10];
    const float* g1    = (const float*)d_in[11];

    float* out_seq   = (float*)d_out;                       // (B, T, H)
    float* out_hlast = out_seq + (size_t)B * T * H;         // (2, B, H)

    char* wp = (char*)d_ws;
    auto carve = [&](size_t bytes) -> void* {
        void* p = (void*)wp;
        wp += (bytes + 255) & ~(size_t)255;
        return p;
    };
    u16*   xbf    = (u16*)  carve((size_t)B * T * DIN * 2);   // 32 MB
    u16*   wih0bf = (u16*)  carve((size_t)H3 * DIN * 2);
    u16*   whh0bf = (u16*)  carve((size_t)H3 * H * 2);
    u16*   wih1bf = (u16*)  carve((size_t)H3 * H * 2);
    u16*   whh1bf = (u16*)  carve((size_t)H3 * H * 2);
    float* gi     = (float*)carve((size_t)B * T * H3 * 4);    // 384 MB, GI0/GI1
    u16*   h0seq  = (u16*)  carve((size_t)B * T * H * 2);     // 64 MB
    float* gh     = (float*)carve((size_t)B * H3 * 4);
    float* hf0    = (float*)carve((size_t)B * H * 4);
    float* hf1    = (float*)carve((size_t)B * H * 4);
    u16*   hbf0   = (u16*)  carve((size_t)B * H * 2);
    u16*   hbf1   = (u16*)  carve((size_t)B * H * 2);
    float* hf[2]  = {hf0, hf1};
    u16*   hbf[2] = {hbf0, hbf1};

    // ---- one-time conversions to bf16 ----
    {
        size_t n = (size_t)B * T * DIN;
        convert_bf16<<<(n + 255) / 256, 256, 0, stream>>>(x, xbf, n);
        n = (size_t)H3 * DIN;
        convert_bf16<<<(n + 255) / 256, 256, 0, stream>>>(w_ih0, wih0bf, n);
        n = (size_t)H3 * H;
        convert_bf16<<<(n + 255) / 256, 256, 0, stream>>>(w_hh0, whh0bf, n);
        convert_bf16<<<(n + 255) / 256, 256, 0, stream>>>(w_ih1, wih1bf, n);
        convert_bf16<<<(n + 255) / 256, 256, 0, stream>>>(w_hh1, whh1bf, n);
    }

    const dim3 blk(128);
    const dim3 grid_big(B * T / 64, H3 / 64);   // 512 x 48
    const dim3 grid_rec(H3 / 64);               // 48

    // ==== Layer 0 ====
    gemm_bf16_wmma<<<grid_big, blk, 0, stream>>>(xbf, wih0bf, b_ih0, gi, B * T, H3, DIN);
    init_state<<<(B * H + 255) / 256, 256, 0, stream>>>(h0, hf[0], hbf[0], B * H);

    int cur = 0;
    for (int t = 0; t < T; ++t) {
        gru_hh_gemm<<<grid_rec, blk, 0, stream>>>(hbf[cur], whh0bf, b_hh0, gh, H3, H);
        gru_gate_rmsnorm<<<B, 256, 0, stream>>>(
            gi + (size_t)t * H3, (size_t)T * H3, gh, hf[cur], g0,
            hf[1 - cur], hbf[1 - cur],
            nullptr, h0seq + (size_t)t * H, (size_t)T * H, H);
        cur ^= 1;
    }
    (void)hipMemcpyAsync(out_hlast, hf[cur], (size_t)B * H * 4,
                         hipMemcpyDeviceToDevice, stream);

    // ==== Layer 1 ====
    gemm_bf16_wmma<<<grid_big, blk, 0, stream>>>(h0seq, wih1bf, b_ih1, gi, B * T, H3, H);
    init_state<<<(B * H + 255) / 256, 256, 0, stream>>>(h0 + (size_t)B * H, hf[0], hbf[0], B * H);

    cur = 0;
    for (int t = 0; t < T; ++t) {
        gru_hh_gemm<<<grid_rec, blk, 0, stream>>>(hbf[cur], whh1bf, b_hh1, gh, H3, H);
        gru_gate_rmsnorm<<<B, 256, 0, stream>>>(
            gi + (size_t)t * H3, (size_t)T * H3, gh, hf[cur], g1,
            hf[1 - cur], hbf[1 - cur],
            out_seq + (size_t)t * H, nullptr, (size_t)T * H, H);
        cur ^= 1;
    }
    (void)hipMemcpyAsync(out_hlast + (size_t)B * H, hf[cur], (size_t)B * H * 4,
                         hipMemcpyDeviceToDevice, stream);
}
